// TDAFrequencyDecomp_66752381714564
// MI455X (gfx1250) — matrix operand order repaired and checked by reading, not compiled, not run on hardware
//
#include <hip/hip_runtime.h>
#include <math.h>
#include <stdint.h>

typedef __attribute__((ext_vector_type(16))) _Float16 v16h;
typedef __attribute__((ext_vector_type(8)))  float    v8f;
typedef __attribute__((ext_vector_type(4)))  unsigned int u32x4;
typedef __attribute__((ext_vector_type(4)))  int      i32x4;
typedef __attribute__((ext_vector_type(8)))  int      i32x8;

#define NTHR 256
#define DM 512        // D_MODEL
#define BB 32         // batch
#define T0 2048       // level0 length

#if defined(__has_builtin)
#if __has_builtin(__builtin_amdgcn_tensor_load_to_lds) && __has_builtin(__builtin_amdgcn_s_wait_tensorcnt)
#define USE_TDM 1
#endif
#endif
#ifndef USE_TDM
#define USE_TDM 0
#endif

__device__ __forceinline__ int bitrev(int x, int bits) {
  return (int)(__brev((unsigned)x) >> (32 - bits));
}

#if USE_TDM
// Issue one TDM descriptor: load a (2 cols x rows) f32 tile whose rows are
// rowStrideElems apart in global memory, packed contiguously into LDS.
// Two floats per row == one float2 slot -> packed complex signal, natural order.
__device__ __forceinline__ void tdm_load_2col(const float* gbase, unsigned lds_byte_addr,
                                              int rows, int rowStrideElems) {
  const unsigned long long ga = (unsigned long long)(uintptr_t)gbase;
  u32x4 g0;
  g0[0] = 1u;                                            // count=1 (valid), user mode
  g0[1] = lds_byte_addr;                                 // lds_addr
  g0[2] = (unsigned)(ga & 0xffffffffull);                // global_addr[31:0]
  g0[3] = (unsigned)((ga >> 32) & 0x1ffffffull) | (2u << 30); // addr[56:32], type=2
  i32x8 g1;
  g1[0] = (int)(2u << 16);                               // data_size=4B; mask=0
  g1[1] = (int)(2u << 16);                               // tensor_dim0 = 2 (bits 63:48)
  g1[2] = (int)(((unsigned)rows & 0xffffu) << 16);       // tensor_dim1 lo16 (bits 95:80)
  g1[3] = (int)(((unsigned)rows >> 16) & 0xffffu) | (int)(2u << 16); // dim1 hi16, tile_dim0=2
  g1[4] = (int)((unsigned)rows & 0xffffu);               // tile_dim1 = rows; tile_dim2=0
  g1[5] = rowStrideElems;                                // tensor_dim0_stride lo32
  g1[6] = 0;                                             // stride hi16, dim1_stride lo16
  g1[7] = 0;
  const i32x4 z4 = {0, 0, 0, 0};
#if __has_include(<hip/amd_detail/amd_gfx1250_TDM.h>)
  const i32x8 z8 = {0, 0, 0, 0, 0, 0, 0, 0};
  __builtin_amdgcn_tensor_load_to_lds(g0, g1, z4, z4, z8, 0);
#else
  __builtin_amdgcn_tensor_load_to_lds(g0, g1, z4, z4, 0);
#endif
}
#endif

// In-place radix-2 DIT FFT over PAIRS complex LDS buffers.
// Input bit-reversed, output natural. dir=-1 fwd, dir=+1 inverse (unscaled).
template<int PAIRS, int STRIDE>
__device__ __forceinline__ void fft_pairs(float2* buf, int L, int logL, float dir, int tid) {
  const int halfL = L >> 1;
  const int NB = PAIRS * halfL;
  for (int s = 1; s <= logL; ++s) {
    const int half = 1 << (s - 1);
    const int len = half << 1;
    for (int i = tid; i < NB; i += NTHR) {
      const int p   = i / halfL;
      const int j   = i - p * halfL;
      const int grp = j / half;
      const int pos = j - grp * half;
      float2* base = buf + p * STRIDE + grp * len + pos;
      const float ang = dir * 6.28318530717958647692f * (float)pos / (float)len;
      float sn, cs;
      __sincosf(ang, &sn, &cs);
      const float2 u = base[0];
      const float2 v = base[half];
      const float2 t = make_float2(v.x * cs - v.y * sn, v.x * sn + v.y * cs);
      base[0]    = make_float2(u.x + t.x, u.y + t.y);
      base[half] = make_float2(u.x - t.x, u.y - t.y);
    }
    __syncthreads();
  }
}

// In-place bit-reversal permutation (each unordered pair swapped by one thread).
template<int PAIRS, int STRIDE>
__device__ __forceinline__ void bitrev_permute(float2* buf, int L, int logL, int tid) {
  for (int i = tid; i < PAIRS * L; i += NTHR) {
    const int p = i / L;
    const int k = i % L;
    const int r = bitrev(k, logL);
    if (r > k) {
      const float2 t0 = buf[p * STRIDE + k];
      buf[p * STRIDE + k] = buf[p * STRIDE + r];
      buf[p * STRIDE + r] = t0;
    }
  }
  __syncthreads();
}

// ---------------------------------------------------------------------------
// Kernel 1: per-(b,d) TDA stats + 2048-pt FFT spectral centroid.
// WG: one b, 4 consecutive d (2 real channels packed per complex FFT).
// feats[b][d][7] = mean,std,min,max,trend,autocorr,centroid
// ---------------------------------------------------------------------------
__global__ void __launch_bounds__(NTHR)
k_tda_stats(const float* __restrict__ x, float* __restrict__ feats) {
  __shared__ float2 cbuf[4096];   // 2 pairs x 2048 complex = 32 KB
  __shared__ float  red[256];
  const int tid = threadIdx.x;
  const int b   = blockIdx.y;
  const int d0  = blockIdx.x * 4;
  const float* xb = x + (size_t)b * T0 * DM + d0;

#if USE_TDM
  // TDM: two (2 x 2048) tiles land as packed complex, natural order.
  if (tid < 32) {
    for (int p = 0; p < 2; ++p)
      tdm_load_2col(xb + 2 * p, (unsigned)(uintptr_t)(&cbuf[p * 2048]), T0, DM);
    __builtin_amdgcn_s_wait_tensorcnt(0);
  }
  __syncthreads();
#else
  for (int i = tid; i < 4 * T0; i += NTHR) {
    const int dloc = i & 3;
    const int t = i >> 2;
    const float v = xb[(size_t)t * DM + dloc];
    ((float*)&cbuf[(dloc >> 1) * 2048 + t])[dloc & 1] = v;
  }
  __syncthreads();
#endif

  // ---- time-domain stats (natural order in LDS) ----
  const int dloc = tid & 3;
  const int tsub = tid >> 2;      // 64 threads per channel
  auto ldval = [&](int t) -> float {
    return ((float*)&cbuf[(dloc >> 1) * 2048 + t])[dloc & 1];
  };
  float S = 0.f, SQ = 0.f, ST = 0.f, LAG = 0.f;
  float MN = 3.4e38f, MX = -3.4e38f;
  for (int t = tsub; t < T0; t += 64) {
    const float v = ldval(t);
    S += v; SQ += v * v; ST += (float)t * v;
    MN = fminf(MN, v); MX = fmaxf(MX, v);
    if (t < T0 - 1) LAG += v * ldval(t + 1);
  }
  const float x0 = ldval(0);
  const float xl = ldval(T0 - 1);

  auto red_sum = [&](float v) -> float {
    red[dloc * 64 + tsub] = v; __syncthreads();
    for (int st = 32; st > 0; st >>= 1) {
      if (tsub < st) red[dloc * 64 + tsub] += red[dloc * 64 + tsub + st];
      __syncthreads();
    }
    float r = red[dloc * 64]; __syncthreads(); return r;
  };
  auto red_min = [&](float v) -> float {
    red[dloc * 64 + tsub] = v; __syncthreads();
    for (int st = 32; st > 0; st >>= 1) {
      if (tsub < st) red[dloc*64+tsub] = fminf(red[dloc*64+tsub], red[dloc*64+tsub+st]);
      __syncthreads();
    }
    float r = red[dloc * 64]; __syncthreads(); return r;
  };
  auto red_max = [&](float v) -> float {
    red[dloc * 64 + tsub] = v; __syncthreads();
    for (int st = 32; st > 0; st >>= 1) {
      if (tsub < st) red[dloc*64+tsub] = fmaxf(red[dloc*64+tsub], red[dloc*64+tsub+st]);
      __syncthreads();
    }
    float r = red[dloc * 64]; __syncthreads(); return r;
  };

  S = red_sum(S); SQ = red_sum(SQ); ST = red_sum(ST); LAG = red_sum(LAG);
  MN = red_min(MN); MX = red_max(MX);

  if (tsub == 0) {
    float* f = feats + ((size_t)b * DM + d0 + dloc) * 7;
    const float T = (float)T0;
    const float mean = S / T;
    const float var1 = SQ - S * S / T;               // sum((x-mean)^2)
    f[0] = mean;
    f[1] = sqrtf(fmaxf(var1, 0.f) / (T - 1.f));
    f[2] = MN;
    f[3] = MX;
    const float tnum = ST - 1023.5f * S;             // sum(xc*ic)
    const float tden = sqrtf(var1 * 715827712.0f);   // sum(ic^2)=T(T^2-1)/12
    f[4] = (tden > 0.f) ? tnum / tden : 0.f;
    const float n1 = T - 1.f;
    const float Sa = S - xl, Sb = S - x0;
    const float anum = LAG - Sa * Sb / n1;
    const float aden = sqrtf((SQ - xl * xl - Sa * Sa / n1) *
                             (SQ - x0 * x0 - Sb * Sb / n1));
    f[5] = (aden > 0.f) ? anum / aden : 0.f;
  }
  __syncthreads();

  // ---- forward FFT: bit-reverse then DIT -> Y natural order ----
  bitrev_permute<2, 2048>(cbuf, 2048, 11, tid);
  fft_pairs<2, 2048>(cbuf, 2048, 11, -1.0f, tid);

  // ---- spectral centroid over bins 0..1023 ----
  const int p  = tid & 1;
  const int k0 = tid >> 1;        // 128 threads per pair
  float sA = 0.f, sB = 0.f, kA = 0.f, kB = 0.f;
  for (int k = k0; k < 1024; k += 128) {
    const float2 v = cbuf[p * 2048 + k];
    const float2 w = cbuf[p * 2048 + ((2048 - k) & 2047)];
    const float ar = 0.5f * (v.x + w.x), ai = 0.5f * (v.y - w.y);
    const float br = 0.5f * (v.y + w.y), bi = 0.5f * (w.x - v.x);
    const float pa = ar * ar + ai * ai;
    const float pb = br * br + bi * bi;
    sA += pa; sB += pb; kA += (float)k * pa; kB += (float)k * pb;
  }
  auto red_spec = [&](float v) -> float {
    red[p * 128 + k0] = v; __syncthreads();
    for (int st = 64; st > 0; st >>= 1) {
      if (k0 < st) red[p*128+k0] += red[p*128+k0+st];
      __syncthreads();
    }
    float r = red[p * 128]; __syncthreads(); return r;
  };
  sA = red_spec(sA); kA = red_spec(kA); sB = red_spec(sB); kB = red_spec(kB);
  if (k0 == 0) {
    feats[((size_t)b * DM + d0 + 2 * p + 0) * 7 + 6] = (sA > 0.f) ? kA / sA : 0.f;
    feats[((size_t)b * DM + d0 + 2 * p + 1) * 7 + 6] = (sB > 0.f) ? kB / sB : 0.f;
  }
}

// ---------------------------------------------------------------------------
// Kernel 2: tda[b][j] = mean over d of feats[b][d][j]
// ---------------------------------------------------------------------------
__global__ void __launch_bounds__(NTHR)
k_tda_reduce(const float* __restrict__ feats, float* __restrict__ tda) {
  __shared__ float red[256];
  const int b = blockIdx.x;
  for (int j = 0; j < 7; ++j) {
    float s = 0.f;
    for (int d = threadIdx.x; d < DM; d += NTHR)
      s += feats[((size_t)b * DM + d) * 7 + j];
    red[threadIdx.x] = s; __syncthreads();
    for (int st = 128; st > 0; st >>= 1) {
      if (threadIdx.x < st) red[threadIdx.x] += red[threadIdx.x + st];
      __syncthreads();
    }
    if (threadIdx.x == 0) tda[b * 7 + j] = red[0] * (1.0f / (float)DM);
    __syncthreads();
  }
}

// ---------------------------------------------------------------------------
// Kernel 3: h = relu(tda@w1+b1); inj_l = 0.3*(tda@wi_l+bi_l), l=0..2 (wi3 dead)
// ---------------------------------------------------------------------------
__global__ void __launch_bounds__(NTHR)
k_mlp_small(const float* __restrict__ tda,
            const float* __restrict__ w1, const float* __restrict__ b1,
            const float* __restrict__ wi0, const float* __restrict__ bi0,
            const float* __restrict__ wi1, const float* __restrict__ bi1,
            const float* __restrict__ wi2, const float* __restrict__ bi2,
            float* __restrict__ h, float* __restrict__ inj) {
  const int gid = blockIdx.x * NTHR + threadIdx.x;   // 0..16383
  const int b = gid >> 9, n = gid & 511;
  float t[7];
  #pragma unroll
  for (int j = 0; j < 7; ++j) t[j] = tda[b * 7 + j];
  float acc = b1[n];
  #pragma unroll
  for (int j = 0; j < 7; ++j) acc += t[j] * w1[j * DM + n];
  h[gid] = fmaxf(acc, 0.f);
  const float* W[3]  = { wi0, wi1, wi2 };
  const float* Bs[3] = { bi0, bi1, bi2 };
  for (int l = 0; l < 3; ++l) {
    float a = Bs[l][n];
    #pragma unroll
    for (int j = 0; j < 7; ++j) a += t[j] * W[l][j * DM + n];
    inj[(size_t)l * (BB * DM) + gid] = 0.3f * a;
  }
}

// ---------------------------------------------------------------------------
// Kernel 4: attn = sigmoid(h @ w2 + b2) via V_WMMA_F32_16X16X32_F16.
// ---------------------------------------------------------------------------
__global__ void __launch_bounds__(NTHR)
k_attn_wmma(const float* __restrict__ h, const float* __restrict__ w2,
            const float* __restrict__ b2, float* __restrict__ attn) {
  const int wave = threadIdx.x >> 5;
  const int lane = threadIdx.x & 31;
  const int tileId = blockIdx.x * 8 + wave;  // 0..63
  const int m0 = (tileId >> 5) * 16;
  const int n0 = (tileId & 31) * 16;
  const int l16 = lane & 15;
  const int hi  = lane >> 4;
  v8f acc = {};
  for (int kk = 0; kk < DM; kk += 32) {
    v16h a, bf;
    #pragma unroll
    for (int e = 0; e < 16; ++e) {
      const int ka = (e & 7) + ((e >> 3) << 4) + hi * 8;   // A 16x32 layout
      a[e] = (_Float16)h[(m0 + l16) * DM + kk + ka];
      const int kb = (e & 7) + ((e >> 3) << 4) + hi * 8;   // B 32x16 layout
      bf[e] = (_Float16)w2[(size_t)(kk + kb) * DM + n0 + l16];
    }
    acc = __builtin_amdgcn_wmma_f32_16x16x32_f16(false, a, false, bf,
                                                 (short)0, acc, false, false);
  }
  const int ncol = n0 + l16;
  #pragma unroll
  for (int r = 0; r < 8; ++r) {
    const int brow = m0 + hi * 8 + r;
    const float v = acc[r] + b2[ncol];
    attn[brow * DM + ncol] = 1.0f / (1.0f + __expf(-v));
  }
}

// ---------------------------------------------------------------------------
// Kernel 5 (x3): fused frequency-domain 2x upsample + combine.
// res = irfft(zeropad(rfft(src)), 2N) * 0.5 ; out = (high - res)*attn + inj
// TDM stages each packed-complex channel pair into LDS; L2 warmed for 'high'.
// ---------------------------------------------------------------------------
template<int LOGN, int PAIRS>
__global__ void __launch_bounds__(NTHR)
k_upsample(const float* __restrict__ src,   // (B, N, 512)
           const float* __restrict__ high,  // (B, 2N, 512)
           const float* __restrict__ attn,  // (B, 512)
           const float* __restrict__ inj,   // (B, 512)
           float* __restrict__ out) {       // (B, 2N, 512)
  constexpr int N  = 1 << LOGN;
  constexpr int L2 = N * 2;
  constexpr int DW = PAIRS * 2;             // d channels per WG
  __shared__ float2 cbuf[4096];             // PAIRS * L2 == 4096 -> 32 KB
  const int tid = threadIdx.x;
  const int b   = blockIdx.y;
  const int d0  = blockIdx.x * DW;
  const float* sb = src + (size_t)b * N * DM + d0;
  const float* hb = high + (size_t)b * L2 * DM + d0;

#if USE_TDM
  // A: TDM-load PAIRS (2 x N) tiles -> packed complex, natural order.
  if (tid < 32) {
    #pragma unroll
    for (int p = 0; p < PAIRS; ++p)
      tdm_load_2col(sb + 2 * p, (unsigned)(uintptr_t)(&cbuf[p * L2]), N, DM);
    __builtin_amdgcn_s_wait_tensorcnt(0);
  }
  // warm L2 for the strided 'high' stream consumed in the combine phase
  for (int m = tid; m < L2; m += NTHR)
    __builtin_prefetch(hb + (size_t)m * DM, 0, 1);
  __syncthreads();
  bitrev_permute<PAIRS, L2>(cbuf, N, LOGN, tid);
#else
  for (int i = tid; i < DW * N; i += NTHR) {
    const int dloc = i % DW;
    const int t = i / DW;
    const float v = sb[(size_t)t * DM + dloc];
    ((float*)&cbuf[(dloc >> 1) * L2 + bitrev(t, LOGN)])[dloc & 1] = v;
  }
  __syncthreads();
#endif

  // B: forward FFT length N -> Y in [0,N)
  fft_pairs<PAIRS, L2>(cbuf, N, LOGN, -1.0f, tid);

  // C: expand Hermitian spectrum to 2N with zero-padded middle:
  //    C2[0..N/2]=Y[0..N/2]; C2[3N/2..2N-1]=Y[N/2..N-1]; zeros between.
  for (int i = tid; i < PAIRS * (N / 2); i += NTHR) {
    const int p = i / (N / 2);
    const int j = i % (N / 2);
    cbuf[p * L2 + N + N / 2 + j] = cbuf[p * L2 + N / 2 + j];
  }
  __syncthreads();
  for (int i = tid; i < PAIRS * (N - 1); i += NTHR) {
    const int p = i / (N - 1);
    const int j = i % (N - 1);
    cbuf[p * L2 + N / 2 + 1 + j] = make_float2(0.f, 0.f);
  }
  __syncthreads();

  // D: bit-reverse permute length 2N, then inverse FFT (unscaled)
  bitrev_permute<PAIRS, L2>(cbuf, L2, LOGN + 1, tid);
  fft_pairs<PAIRS, L2>(cbuf, L2, LOGN + 1, +1.0f, tid);

  // E: combine + store: scale = (1/(2N) ifft norm) * (src/tgt = 0.5)
  const float scale = 1.0f / (4.0f * (float)N);
  float* ob = out + (size_t)b * L2 * DM + d0;
  for (int i = tid; i < DW * L2; i += NTHR) {
    const int dloc = i % DW;
    const int m = i / DW;
    const float res = ((const float*)&cbuf[(dloc >> 1) * L2 + m])[dloc & 1] * scale;
    const float hv = hb[(size_t)m * DM + dloc];
    const float o = (hv - res) * attn[b * DM + d0 + dloc] + inj[b * DM + d0 + dloc];
    ob[(size_t)m * DM + dloc] = o;
  }
}

// ---------------------------------------------------------------------------
extern "C" void kernel_launch(void* const* d_in, const int* in_sizes, int n_in,
                              void* d_out, int out_size, void* d_ws, size_t ws_size,
                              hipStream_t stream) {
  (void)in_sizes; (void)n_in; (void)out_size; (void)ws_size;
  const float* level0 = (const float*)d_in[0];
  const float* level1 = (const float*)d_in[1];
  const float* level2 = (const float*)d_in[2];
  const float* level3 = (const float*)d_in[3];
  const float* w1  = (const float*)d_in[4];
  const float* b1  = (const float*)d_in[5];
  const float* w2  = (const float*)d_in[6];
  const float* b2  = (const float*)d_in[7];
  const float* wi0 = (const float*)d_in[8];
  const float* bi0 = (const float*)d_in[9];
  const float* wi1 = (const float*)d_in[10];
  const float* bi1 = (const float*)d_in[11];
  const float* wi2 = (const float*)d_in[12];
  const float* bi2 = (const float*)d_in[13];
  float* out = (float*)d_out;
  float* ws  = (float*)d_ws;

  float* feats = ws;                                // 32*512*7 = 114688
  float* tda   = feats + (size_t)BB * DM * 7;       // 224
  float* h     = tda + BB * 7;                      // 16384
  float* attn  = h + (size_t)BB * DM;               // 16384
  float* inj   = attn + (size_t)BB * DM;            // 3*16384

  k_tda_stats<<<dim3(DM / 4, BB), NTHR, 0, stream>>>(level0, feats);
  k_tda_reduce<<<BB, NTHR, 0, stream>>>(feats, tda);
  k_mlp_small<<<(BB * DM) / NTHR, NTHR, 0, stream>>>(tda, w1, b1, wi0, bi0,
                                                     wi1, bi1, wi2, bi2, h, inj);
  k_attn_wmma<<<8, NTHR, 0, stream>>>(h, w2, b2, attn);

  const size_t o1 = (size_t)BB * 2048 * DM;
  const size_t o2 = o1 + (size_t)BB * 1024 * DM;
  const size_t o3 = o2 + (size_t)BB * 512 * DM;

  k_upsample<10, 2><<<dim3(DM / 4, BB), NTHR, 0, stream>>>(
      level1, level0, attn, inj + 0 * (BB * DM), out);
  k_upsample<9, 4><<<dim3(DM / 8, BB), NTHR, 0, stream>>>(
      level2, level1, attn, inj + 1 * (BB * DM), out + o1);
  k_upsample<8, 8><<<dim3(DM / 16, BB), NTHR, 0, stream>>>(
      level3, level2, attn, inj + 2 * (BB * DM), out + o2);

  hipMemcpyAsync(out + o3, level3, (size_t)BB * 256 * DM * sizeof(float),
                 hipMemcpyDeviceToDevice, stream);
}
